// MetaLa_70042326663979
// MI455X (gfx1250) — matrix operand order repaired and checked by reading, not compiled, not run on hardware
//
#include <hip/hip_runtime.h>

#define DEV __device__ __forceinline__

typedef __attribute__((ext_vector_type(16))) __bf16 v16bf;
typedef __attribute__((ext_vector_type(8)))  float  v8f;
typedef __attribute__((ext_vector_type(4)))  int    v4i;

#if __has_builtin(__builtin_amdgcn_global_load_async_to_lds_b128) && \
    __has_builtin(__builtin_amdgcn_s_wait_asynccnt)
#define USE_ASYNC_LDS 1
#else
#define USE_ASYNC_LDS 0
#endif

#if USE_ASYNC_LDS
typedef __attribute__((address_space(1))) v4i* as1_v4i;
typedef __attribute__((address_space(3))) v4i* as3_v4i;
// builtin signature (probe-verified): (v4i __device__*, v4i __shared__*, imm, imm)
#define ASYNC_B128(gp, lp)                                              \
  __builtin_amdgcn_global_load_async_to_lds_b128((as1_v4i)(v4i*)(gp),   \
                                                 (as3_v4i)(v4i*)(lp), 0, 0)
#endif

// problem constants
constexpr int Bb = 2, Nn = 4096, Ee = 1024, Dh = 128, Hh = 8;
constexpr int Mm = Bb * Nn;       // 8192 token rows
constexpr int CHUNK = 64;         // GLA chunk length
constexpr int TILE_E = 128 * 32;  // elements in one k-stage tile

// dynamic LDS carve for the scan kernel (vv is double-buffered)
constexpr int SM_S    = Dh * Dh * 4;           // 65536  fp32 state
constexpr int SM_QKV  = 5 * CHUNK * Dh * 2;    // 81920  qhat/qk/ktil/vv0/vv1 bf16
constexpr int SM_GC   = CHUNK * Dh * 4;        // 32768  fp32 cumulative gates
constexpr int SM_A    = CHUNK * 72 * 2;        // 9216   bf16 intra attention (ld=72)
constexpr int SM_EBC  = 2 * Dh * 4;            // 1024   exp(b_C) and b_C
constexpr int SMEM_SCAN = SM_S + SM_QKV + SM_GC + SM_A + SM_EBC;   // 190464

DEV int laneid() { return threadIdx.x & 31; }
// fast sigmoid: one v_exp + one v_rcp (no IEEE-div fixup chain)
DEV float sigf(float x) { return __builtin_amdgcn_rcpf(1.f + __expf(-x)); }

DEV v8f zero8() {
  v8f z;
#pragma unroll
  for (int j = 0; j < 8; ++j) z[j] = 0.f;
  return z;
}

DEV v8f wmma_bf16(v16bf a, v16bf b, v8f c) {
  // D = A(16x32 bf16) * B(32x16 bf16) + C(16x16 f32)
  return __builtin_amdgcn_wmma_f32_16x16x32_bf16(false, a, false, b, (short)0, c,
                                                 false, false);
}

// ---- fragment loaders (per CDNA5 ISA 7.12.2 VGPR layouts, wave32) ----

// A(m,k) = base[r*ld + k], 16x32 bf16 tile
DEV v16bf frag_a_row(const __bf16* base, int ld) {
  const int lane = laneid();
  const int r = lane & 15, kh = (lane >> 4) * 8;
  const __bf16* p = base + r * ld;
  v16bf a;
#pragma unroll
  for (int j = 0; j < 8; ++j) {
    int k = ((j < 4) ? 2 * j : 16 + 2 * (j - 4)) + kh;
    a[2 * j] = p[k]; a[2 * j + 1] = p[k + 1];
  }
  return a;
}

// A(m,k) from fp32 LDS tile, convert on read
DEV v16bf frag_a_row_f32(const float* base, int ld) {
  const int lane = laneid();
  const int r = lane & 15, kh = (lane >> 4) * 8;
  const float* p = base + r * ld;
  v16bf a;
#pragma unroll
  for (int j = 0; j < 8; ++j) {
    int k = ((j < 4) ? 2 * j : 16 + 2 * (j - 4)) + kh;
    a[2 * j] = (__bf16)p[k]; a[2 * j + 1] = (__bf16)p[k + 1];
  }
  return a;
}

// A(m,k) = base[k*ld + m]  (transposed source, used for K~^T)
DEV v16bf frag_a_trans(const __bf16* base, int ld) {
  const int lane = laneid();
  const int r = lane & 15, kh = (lane >> 4) * 8;
  v16bf a;
#pragma unroll
  for (int j = 0; j < 8; ++j) {
    int k = ((j < 4) ? 2 * j : 16 + 2 * (j - 4)) + kh;
    a[2 * j]     = base[k * ld + r];
    a[2 * j + 1] = base[(k + 1) * ld + r];
  }
  return a;
}

// B(k,c) = base[c*ld + k]  (B^T source stored row-major per output column)
DEV v16bf frag_b_col(const __bf16* base, int ld) {
  const int lane = laneid();
  const int c = lane & 15, ko = (lane >> 4) * 8;
  const __bf16* p = base + c * ld + ko;
  v16bf b;
#pragma unroll
  for (int j = 0; j < 8; ++j) { int k = (j < 4) ? 2 * j : 2 * j + 8; b[2 * j] = p[k]; b[2 * j + 1] = p[k + 1]; }
  return b;
}

// B(k,c) from fp32 LDS tile laid out [c][k], convert on read
DEV v16bf frag_b_col_f32(const float* base, int ld) {
  const int lane = laneid();
  const int c = lane & 15, ko = (lane >> 4) * 8;
  const float* p = base + c * ld + ko;
  v16bf b;
#pragma unroll
  for (int j = 0; j < 8; ++j) {
    int k = (j < 4) ? 2 * j : 2 * j + 8;
    b[2 * j] = (__bf16)p[k]; b[2 * j + 1] = (__bf16)p[k + 1];
  }
  return b;
}

// B(k,c) = base[k*ld + c]  (row-major B)
DEV v16bf frag_b_row(const __bf16* base, int ld) {
  const int lane = laneid();
  const int c = lane & 15, ko = (lane >> 4) * 8;
  v16bf b;
#pragma unroll
  for (int j = 0; j < 8; ++j) {
    int k = ko + ((j < 4) ? 2 * j : 2 * j + 8);
    b[2 * j]     = base[k * ld + c];
    b[2 * j + 1] = base[(k + 1) * ld + c];
  }
  return b;
}

// B(k,c) from fp32 source with on-the-fly bf16 convert (state matrix S)
DEV v16bf frag_b_row_f32(const float* base, int ld) {
  const int lane = laneid();
  const int c = lane & 15, ko = (lane >> 4) * 8;
  v16bf b;
#pragma unroll
  for (int j = 0; j < 8; ++j) {
    int k = ko + ((j < 4) ? 2 * j : 2 * j + 8);
    b[2 * j]     = (__bf16)base[k * ld + c];
    b[2 * j + 1] = (__bf16)base[(k + 1) * ld + c];
  }
  return b;
}

// ---- shared 128x128-tile GEMM core, double-buffered async LDS pipeline ----
// A: [M x K] row-major (f32 or bf16).  W: [Ncols x K] row-major f32 (output col = W row).
// lA_/lB hold TWO k-stage tiles each (source dtype); frags convert on read.
template <bool ABF16>
DEV void gemm_core(const void* Ag, const float* Wg, int K, int row0, int col0,
                   v8f acc[2][4], void* lA_, float* lB) {
  const int tid = threadIdx.x;
  const int wid = tid >> 5;
  const int wm = wid & 3, wn = wid >> 2;   // wave covers rows [wm*32,+32), cols [wn*64,+64)
#pragma unroll
  for (int mi = 0; mi < 2; ++mi)
#pragma unroll
    for (int ni = 0; ni < 4; ++ni) acc[mi][ni] = zero8();

#if USE_ASYNC_LDS
  constexpr short NFL = ABF16 ? 6 : 8;     // async ops in flight per lane per stage
  auto issue = [&](int kk, int buf) {
    if (ABF16) {
      const __bf16* A = (const __bf16*)Ag;
      __bf16* lA = (__bf16*)lA_ + buf * TILE_E;
#pragma unroll
      for (int i = 0; i < 2; ++i) {        // 128x32 bf16 = 8KB, 16B per lane-op
        int chunk = tid + i * 256;
        int row = chunk >> 2, cb = (chunk & 3) * 8;
        ASYNC_B128(A + (size_t)(row0 + row) * K + kk + cb, lA + row * 32 + cb);
      }
    } else {
      const float* A = (const float*)Ag;
      float* lA = (float*)lA_ + buf * TILE_E;
#pragma unroll
      for (int i = 0; i < 4; ++i) {        // 128x32 f32 = 16KB
        int chunk = tid + i * 256;
        int row = chunk >> 3, cb = (chunk & 7) * 4;
        ASYNC_B128(A + (size_t)(row0 + row) * K + kk + cb, lA + row * 32 + cb);
      }
    }
    float* lBb = lB + buf * TILE_E;
#pragma unroll
    for (int i = 0; i < 4; ++i) {          // W tile 128x32 f32 = 16KB
      int chunk = tid + i * 256;
      int row = chunk >> 3, cb = (chunk & 7) * 4;
      ASYNC_B128(Wg + (size_t)(col0 + row) * K + kk + cb, lBb + row * 32 + cb);
    }
  };
  issue(0, 0);
#endif

  const int nsteps = K >> 5;
  for (int s = 0; s < nsteps; ++s) {
    const int buf = s & 1;
#if USE_ASYNC_LDS
    if (s + 1 < nsteps) {
      issue((s + 1) * 32, (s + 1) & 1);    // prefetch next stage into other buffer
      __builtin_amdgcn_s_wait_asynccnt(NFL);  // current stage landed; next in flight
    } else {
      __builtin_amdgcn_s_wait_asynccnt(0);
    }
#else
    {
      const int kk = s * 32;
      const int r = tid >> 1, kb = (tid & 1) * 16;
      if (ABF16) {
        const __bf16* A = (const __bf16*)Ag;
        __bf16* lA = (__bf16*)lA_ + buf * TILE_E;
        const __bf16* src = A + (size_t)(row0 + r) * K + kk + kb;
#pragma unroll
        for (int i = 0; i < 16; ++i) lA[r * 32 + kb + i] = src[i];
      } else {
        const float* A = (const float*)Ag;
        float* lA = (float*)lA_ + buf * TILE_E;
        const float* src = A + (size_t)(row0 + r) * K + kk + kb;
#pragma unroll
        for (int i = 0; i < 16; ++i) lA[r * 32 + kb + i] = src[i];
      }
      const float* wsrc = Wg + (size_t)(col0 + r) * K + kk + kb;
#pragma unroll
      for (int i = 0; i < 16; ++i) lB[buf * TILE_E + r * 32 + kb + i] = wsrc[i];
    }
#endif
    __syncthreads();

    v16bf a0, a1;
    if (ABF16) {
      const __bf16* lA = (const __bf16*)lA_ + buf * TILE_E;
      a0 = frag_a_row(lA + (wm * 32) * 32, 32);
      a1 = frag_a_row(lA + (wm * 32 + 16) * 32, 32);
    } else {
      const float* lA = (const float*)lA_ + buf * TILE_E;
      a0 = frag_a_row_f32(lA + (wm * 32) * 32, 32);
      a1 = frag_a_row_f32(lA + (wm * 32 + 16) * 32, 32);
    }
    const float* lBb = lB + buf * TILE_E;
#pragma unroll
    for (int ni = 0; ni < 4; ++ni) {
      v16bf b = frag_b_col_f32(lBb + (wn * 64 + ni * 16) * 32, 32);
      acc[0][ni] = wmma_bf16(a0, b, acc[0][ni]);
      acc[1][ni] = wmma_bf16(a1, b, acc[1][ni]);
    }
    __syncthreads();   // readers done before this buffer is re-filled (stage s+2)
  }
}

// epilogue coordinate helper
#define EPI_COORDS                                              \
  const int tid = threadIdx.x, lane = tid & 31, wid = tid >> 5; \
  const int wm = wid & 3, wn = wid >> 2;

// ---- K1: U = X @ W_in^T with fused activations -> v, silu(q), 1-sigmoid, logsigmoid ----
__global__ void k_proj_in(const float* __restrict__ x, const float* __restrict__ Win,
                          __bf16* __restrict__ qb, __bf16* __restrict__ kb,
                          __bf16* __restrict__ vb, float* __restrict__ gf) {
  __shared__ float lA[2 * TILE_E], lB[2 * TILE_E];
  const int row0 = blockIdx.y * 128, col0 = blockIdx.x * 128;
  v8f acc[2][4];
  gemm_core<false>(x, Win, Ee, row0, col0, acc, lA, lB);
  EPI_COORDS
#pragma unroll
  for (int mi = 0; mi < 2; ++mi)
#pragma unroll
    for (int ni = 0; ni < 4; ++ni)
#pragma unroll
      for (int j = 0; j < 8; ++j) {
        int row = row0 + wm * 32 + mi * 16 + (lane >> 4) * 8 + j;
        int col = col0 + wn * 64 + ni * 16 + (lane & 15);
        float v = acc[mi][ni][j];
        int seg = col >> 10, c = col & 1023;
        size_t idx = (size_t)row * Ee + c;
        if (seg == 0) {
          vb[idx] = (__bf16)v;
        } else if (seg == 1) {
          qb[idx] = (__bf16)(v * sigf(v));                    // silu
        } else {
          kb[idx] = (__bf16)(1.f - sigf(v));                  // k = 1 - f
          gf[idx] = fminf(v, 0.f) - __logf(1.f + __expf(-fabsf(v)));  // logsigmoid
        }
      }
}

// ---- K2: T = X @ W_g1^T  (low-rank gate, N=128) ----
__global__ void k_proj_g1(const float* __restrict__ x, const float* __restrict__ Wg1,
                          __bf16* __restrict__ Tb) {
  __shared__ float lA[2 * TILE_E], lB[2 * TILE_E];
  const int row0 = blockIdx.y * 128;
  v8f acc[2][4];
  gemm_core<false>(x, Wg1, Ee, row0, 0, acc, lA, lB);
  EPI_COORDS
#pragma unroll
  for (int mi = 0; mi < 2; ++mi)
#pragma unroll
    for (int ni = 0; ni < 4; ++ni)
#pragma unroll
      for (int j = 0; j < 8; ++j) {
        int row = row0 + wm * 32 + mi * 16 + (lane >> 4) * 8 + j;
        int col = wn * 64 + ni * 16 + (lane & 15);
        Tb[(size_t)row * Dh + col] = (__bf16)acc[mi][ni][j];
      }
}

// ---- K3: chunked GLA scan, one (batch,head) per workgroup, WMMA for all contractions ----
__global__ void k_gla_scan(const __bf16* __restrict__ qb, const __bf16* __restrict__ kb,
                           const __bf16* __restrict__ vb, const float* __restrict__ gf,
                           float* __restrict__ O) {
  extern __shared__ char smem[];
  float*  S    = (float*)smem;                              // 128x128 fp32 state
  __bf16* qhat = (__bf16*)(smem + SM_S);                    // q * exp(b_t)
  __bf16* qk   = qhat + CHUNK * Dh;                         // q * exp(b_t - b_C)
  __bf16* ktil = qk + CHUNK * Dh;                           // k * exp(b_C - b_t)
  __bf16* vv0  = ktil + CHUNK * Dh;                         // v (double-buffered)
  __bf16* vv1  = vv0 + CHUNK * Dh;
  float*  gc   = (float*)(smem + SM_S + SM_QKV);            // cumulative log gates
  __bf16* Am   = (__bf16*)(smem + SM_S + SM_QKV + SM_GC);   // intra attention, ld=72
  float*  ebC  = (float*)(smem + SM_S + SM_QKV + SM_GC + SM_A);  // exp(b_C)
  float*  bCl  = ebC + Dh;                                  // b_C (log)

  const int tid = threadIdx.x, lane = tid & 31, wid = tid >> 5;
  const int bh = blockIdx.x, bi = bh >> 3, h = bh & 7;
  const size_t row_base = (size_t)bi * Nn;
  const int col_base = h * Dh;

  for (int i = tid; i < Dh * Dh; i += 256) S[i] = 0.f;

#if USE_ASYNC_LDS
  auto issue_v = [&](int c0, __bf16* dst) {   // 64x128 bf16 = 16KB DMA
#pragma unroll
    for (int i = 0; i < 4; ++i) {
      int chunk = tid + i * 256;
      int t = chunk >> 4, cb = (chunk & 15) * 8;
      ASYNC_B128(vb + (row_base + c0 + t) * Ee + col_base + cb, dst + t * Dh + cb);
    }
  };
  issue_v(0, vv0);
#endif
  __syncthreads();

  for (int c0 = 0; c0 < Nn; c0 += CHUNK) {
    const int p = (c0 >> 6) & 1;
    __bf16* vv = p ? vv1 : vv0;
#if USE_ASYNC_LDS
    if (c0 + CHUNK < Nn) issue_v(c0 + CHUNK, p ? vv0 : vv1);
#endif
    // (1) per-feature cumulative log-gates  b_t[d], and b_C / e^{b_C}
    if (tid < Dh) {
      float s = 0.f;
#pragma unroll 1
      for (int t = 0; t < CHUNK; ++t) {
        s += gf[(row_base + c0 + t) * Ee + col_base + tid];
        gc[t * Dh + tid] = s;
      }
      bCl[tid] = s;
      ebC[tid] = __expf(s);
    }
    __syncthreads();

    // (2) build scaled bf16 operand tiles (+ prefetch next chunk's streams)
    for (int idx = tid; idx < CHUNK * Dh; idx += 256) {
      int t = idx >> 7, d = idx & 127;
      size_t g = (row_base + c0 + t) * Ee + col_base + d;
      float bt = gc[idx], bc = bCl[d];
      float q = (float)qb[g], k = (float)kb[g];
      qhat[idx] = (__bf16)(q * __expf(bt));
      qk[idx]   = (__bf16)(q * __expf(bt - bc));
      ktil[idx] = (__bf16)(k * __expf(bc - bt));
#if !USE_ASYNC_LDS
      vv[idx] = vb[g];
#endif
      if (c0 + CHUNK < Nn) {
        size_t gn = g + (size_t)CHUNK * Ee;
        __builtin_prefetch(qb + gn, 0, 1);
        __builtin_prefetch(kb + gn, 0, 1);
        __builtin_prefetch(gf + gn, 0, 1);
      }
    }
#if USE_ASYNC_LDS
    // current V landed; next chunk's V (4 ops) may remain in flight
    if (c0 + CHUNK < Nn) {
      __builtin_amdgcn_s_wait_asynccnt(4);
    } else {
      __builtin_amdgcn_s_wait_asynccnt(0);
    }
#endif
    __syncthreads();

    // (3) intra-chunk attention A[t,s] = sum_d q e^{b_t-b_s} k   (causal)
#pragma unroll
    for (int tt = 0; tt < 2; ++tt) {
      int tile = wid * 2 + tt;
      int ti = tile >> 2, si = tile & 3;
      v8f a = zero8();
#pragma unroll
      for (int ks = 0; ks < 4; ++ks) {
        v16bf af = frag_a_row(qk + (ti * 16) * Dh + ks * 32, Dh);
        v16bf bf = frag_b_col(ktil + (si * 16) * Dh + ks * 32, Dh);
        a = wmma_bf16(af, bf, a);
      }
#pragma unroll
      for (int j = 0; j < 8; ++j) {
        int t = ti * 16 + (lane >> 4) * 8 + j;
        int s = si * 16 + (lane & 15);
        Am[t * 72 + s] = (__bf16)((s <= t) ? a[j] : 0.f);
      }
    }
    __syncthreads();

    // (4) O = A @ V + (q e^{b_t}) @ S_prev
    {
      const int mt = wid & 3;
      const int ng = (wid >> 2) * 4;
      v8f oa[4];
#pragma unroll
      for (int ni = 0; ni < 4; ++ni) oa[ni] = zero8();
#pragma unroll
      for (int ks = 0; ks < 2; ++ks) {
        v16bf af = frag_a_row(Am + (mt * 16) * 72 + ks * 32, 72);
#pragma unroll
        for (int ni = 0; ni < 4; ++ni) {
          v16bf bf = frag_b_row(vv + (ks * 32) * Dh + (ng + ni) * 16, Dh);
          oa[ni] = wmma_bf16(af, bf, oa[ni]);
        }
      }
#pragma unroll
      for (int ks = 0; ks < 4; ++ks) {
        v16bf af = frag_a_row(qhat + (mt * 16) * Dh + ks * 32, Dh);
#pragma unroll
        for (int ni = 0; ni < 4; ++ni) {
          v16bf bf = frag_b_row_f32(S + (ks * 32) * Dh + (ng + ni) * 16, Dh);
          oa[ni] = wmma_bf16(af, bf, oa[ni]);
        }
      }
#pragma unroll
      for (int ni = 0; ni < 4; ++ni)
#pragma unroll
        for (int j = 0; j < 8; ++j) {
          int m = mt * 16 + (lane >> 4) * 8 + j;
          int n = (ng + ni) * 16 + (lane & 15);
          O[(row_base + c0 + m) * Ee + col_base + n] = oa[ni][j];
        }
    }
    __syncthreads();

    // (5) S = diag(e^{b_C}) S + K~^T V   (each wave owns 16 state rows)
    {
      v8f sa[8];
#pragma unroll
      for (int ni = 0; ni < 8; ++ni)
#pragma unroll
        for (int j = 0; j < 8; ++j) {
          int m = wid * 16 + (lane >> 4) * 8 + j;
          int n = ni * 16 + (lane & 15);
          sa[ni][j] = S[m * Dh + n] * ebC[m];
        }
#pragma unroll
      for (int ks = 0; ks < 2; ++ks) {
        v16bf af = frag_a_trans(ktil + (ks * 32) * Dh + wid * 16, Dh);
#pragma unroll
        for (int ni = 0; ni < 8; ++ni) {
          v16bf bf = frag_b_row(vv + (ks * 32) * Dh + ni * 16, Dh);
          sa[ni] = wmma_bf16(af, bf, sa[ni]);
        }
      }
#pragma unroll
      for (int ni = 0; ni < 8; ++ni)
#pragma unroll
        for (int j = 0; j < 8; ++j) {
          int m = wid * 16 + (lane >> 4) * 8 + j;
          int n = ni * 16 + (lane & 15);
          S[m * Dh + n] = sa[ni][j];
        }
    }
    __syncthreads();   // vv[p] reads done before next iteration refills it
  }
}

// ---- K4: O *= sigmoid(T @ W_g2^T)   (in-place gate) ----
__global__ void k_gate(const __bf16* __restrict__ Tb, const float* __restrict__ Wg2,
                       float* __restrict__ O) {
  __shared__ __bf16 lA[2 * TILE_E];
  __shared__ float lB[2 * TILE_E];
  const int row0 = blockIdx.y * 128, col0 = blockIdx.x * 128;
  v8f acc[2][4];
  gemm_core<true>(Tb, Wg2, Dh, row0, col0, acc, lA, lB);
  EPI_COORDS
#pragma unroll
  for (int mi = 0; mi < 2; ++mi)
#pragma unroll
    for (int ni = 0; ni < 4; ++ni)
#pragma unroll
      for (int j = 0; j < 8; ++j) {
        int row = row0 + wm * 32 + mi * 16 + (lane >> 4) * 8 + j;
        int col = col0 + wn * 64 + ni * 16 + (lane & 15);
        size_t idx = (size_t)row * Ee + col;
        O[idx] = O[idx] * sigf(acc[mi][ni][j]);
      }
}

// ---- K5: row-wise LayerNorm -> bf16 operand for the final projection ----
__global__ void k_ln(const float* __restrict__ O, const float* __restrict__ lnw,
                     __bf16* __restrict__ Y) {
  const int row = blockIdx.x, tid = threadIdx.x;
  __shared__ float rs[8], rs2[8], mv[2];
  float s = 0.f, s2 = 0.f;
  for (int c = tid; c < Ee; c += 256) {
    float v = O[(size_t)row * Ee + c];
    s += v; s2 += v * v;
  }
#pragma unroll
  for (int o = 16; o > 0; o >>= 1) {
    s += __shfl_down(s, o, 32);
    s2 += __shfl_down(s2, o, 32);
  }
  if ((tid & 31) == 0) { rs[tid >> 5] = s; rs2[tid >> 5] = s2; }
  __syncthreads();
  if (tid == 0) {
    float a = 0.f, b = 0.f;
#pragma unroll
    for (int i = 0; i < 8; ++i) { a += rs[i]; b += rs2[i]; }
    constexpr float inv = 1.f / (float)Ee;
    float mu = a * inv;
    mv[0] = mu;
    mv[1] = rsqrtf(b * inv - mu * mu + 1e-5f);
  }
  __syncthreads();
  float mu = mv[0], rstd = mv[1];
  for (int c = tid; c < Ee; c += 256)
    Y[(size_t)row * Ee + c] = (__bf16)((O[(size_t)row * Ee + c] - mu) * rstd * lnw[c]);
}

// ---- K6: out = Y @ W_out^T  (fp32 output) ----
__global__ void k_proj_out(const __bf16* __restrict__ Y, const float* __restrict__ Wout,
                           float* __restrict__ out) {
  __shared__ __bf16 lA[2 * TILE_E];
  __shared__ float lB[2 * TILE_E];
  const int row0 = blockIdx.y * 128, col0 = blockIdx.x * 128;
  v8f acc[2][4];
  gemm_core<true>(Y, Wout, Ee, row0, col0, acc, lA, lB);
  EPI_COORDS
#pragma unroll
  for (int mi = 0; mi < 2; ++mi)
#pragma unroll
    for (int ni = 0; ni < 4; ++ni)
#pragma unroll
      for (int j = 0; j < 8; ++j) {
        int row = row0 + wm * 32 + mi * 16 + (lane >> 4) * 8 + j;
        int col = col0 + wn * 64 + ni * 16 + (lane & 15);
        out[(size_t)row * Ee + col] = acc[mi][ni][j];
      }
}

extern "C" void kernel_launch(void* const* d_in, const int* in_sizes, int n_in,
                              void* d_out, int out_size, void* d_ws, size_t ws_size,
                              hipStream_t stream) {
  (void)in_sizes; (void)n_in; (void)out_size; (void)ws_size;
  const float* x    = (const float*)d_in[0];
  const float* Win  = (const float*)d_in[1];
  const float* Wout = (const float*)d_in[2];
  const float* Wg1  = (const float*)d_in[3];
  const float* Wg2  = (const float*)d_in[4];
  const float* lnw  = (const float*)d_in[5];
  float* out = (float*)d_out;

  char* w = (char*)d_ws;
  size_t off = 0;
  auto carve = [&](size_t bytes) -> char* {
    char* p = w + off;
    off += (bytes + 255) & ~(size_t)255;
    return p;
  };
  const size_t ME = (size_t)Mm * Ee;
  __bf16* qb = (__bf16*)carve(ME * 2);          // silu(q), bf16
  __bf16* kb = (__bf16*)carve(ME * 2);          // 1 - sigmoid(lg), bf16
  __bf16* vb = (__bf16*)carve(ME * 2);          // v, bf16
  float*  gf = (float*)carve(ME * 4);           // logsigmoid(lg), fp32
  __bf16* Tb = (__bf16*)carve((size_t)Mm * Dh * 2);  // gate low-rank, bf16
  float*  Of = (float*)carve(ME * 4);           // scan output / gated output, fp32
  __bf16* Yb = (__bf16*)carve(ME * 2);          // post-LN, bf16

  (void)hipFuncSetAttribute((const void*)k_gla_scan,
                            hipFuncAttributeMaxDynamicSharedMemorySize, SMEM_SCAN);

  k_proj_in <<<dim3(3 * Ee / 128, Mm / 128), 256, 0, stream>>>(x, Win, qb, kb, vb, gf);
  k_proj_g1 <<<dim3(1, Mm / 128),            256, 0, stream>>>(x, Wg1, Tb);
  k_gla_scan<<<dim3(Bb * Hh), 256, SMEM_SCAN, stream>>>(qb, kb, vb, gf, Of);
  k_gate    <<<dim3(Ee / 128, Mm / 128),     256, 0, stream>>>(Tb, Wg2, Of);
  k_ln      <<<dim3(Mm),                     256, 0, stream>>>(Of, lnw, Yb);
  k_proj_out<<<dim3(Ee / 128, Mm / 128),     256, 0, stream>>>(Yb, Wout, out);
}
